// SCELoss_326417514731
// MI455X (gfx1250) — matrix operand order repaired and checked by reading, not compiled, not run on hardware
//
#include <hip/hip_runtime.h>
#include <hip/hip_bf16.h>
#include <stdint.h>

#define KROWS 16384
#define DDIM  128

typedef __bf16 bf16;
typedef __attribute__((ext_vector_type(16))) __bf16 v16bf;
typedef __attribute__((ext_vector_type(8)))  float  v8f;

union Frag { v16bf v; uint4 q[2]; };

__device__ __forceinline__ float ex2(float x){
#if __has_builtin(__builtin_amdgcn_exp2f)
  return __builtin_amdgcn_exp2f(x);      // v_exp_f32 (2^x)
#else
  return exp2f(x);
#endif
}
__device__ __forceinline__ float lg2(float x){
#if __has_builtin(__builtin_amdgcn_logf)
  return __builtin_amdgcn_logf(x);       // v_log_f32 (log2)
#else
  return log2f(x);
#endif
}

// fp32 -> bf16 hi/lo split (hi + lo carries ~16 mantissa bits).
__global__ void sce_convert(const float* __restrict__ x,
                            bf16* __restrict__ hi, bf16* __restrict__ lo,
                            float scale, int n){
  int i = blockIdx.x * 256 + threadIdx.x;
  if(i < n){
    float v = x[i] * scale;
    bf16 h = (bf16)v;
    hi[i] = h;
    lo[i] = (bf16)(v - (float)h);
  }
}

// Each block: one 16-row tile. 8 waves each stream two 16-column strips per
// iteration with WMMA bf16 (hi/lo split) + online base-2 log-sum-exp.
__global__ __launch_bounds__(256) void sce_main(
    const bf16* __restrict__ Phi, const bf16* __restrict__ Plo,
    const bf16* __restrict__ Chi, const bf16* __restrict__ Clo,
    const float* __restrict__ parts, const float* __restrict__ centers,
    float* __restrict__ row_loss)
{
  const int lane = threadIdx.x & 31;
  const int wave = threadIdx.x >> 5;
  const int l16  = lane & 15;
  const int half = lane >> 4;
  const int r0   = blockIdx.x * 16;

  // ---- A fragments: 16 rows x 128 K, hi+lo, register resident ----
  // 16-bit A layout: lane(0-15)=M holds K 0-7,16-23; lane(16-31)=M holds K 8-15,24-31.
  v16bf ahi[4], alo[4];
  {
    const bf16* ph = Phi + (size_t)(r0 + l16) * DDIM;
    const bf16* pl = Plo + (size_t)(r0 + l16) * DDIM;
#pragma unroll
    for(int c = 0; c < 4; ++c){
      const int k0 = 32*c + 8*half;
      Frag f;
      f.q[0] = *(const uint4*)(ph + k0);
      f.q[1] = *(const uint4*)(ph + k0 + 16);
      ahi[c] = f.v;
      f.q[0] = *(const uint4*)(pl + k0);
      f.q[1] = *(const uint4*)(pl + k0 + 16);
      alo[c] = f.v;
    }
  }

  // online softmax state (base-2): one (m,s) per row per lane
  float m[8], s[8];
#pragma unroll
  for(int i = 0; i < 8; ++i){ m[i] = -3.0e38f; s[i] = 0.0f; }

  for(int it = 0; it < 64; ++it){
    // wave w owns column tiles (it*16 + 2w) and (it*16 + 2w + 1)
    const int col0 = (it*16 + 2*wave)*16 + l16;
    const bf16* ch0 = Chi + (size_t)col0 * DDIM;
    const bf16* cl0 = Clo + (size_t)col0 * DDIM;
    const bf16* ch1 = ch0 + 16*DDIM;
    const bf16* cl1 = cl0 + 16*DDIM;
    // prefetch next iteration's strips into near cache (WGP scope)
    __builtin_prefetch(ch0 + 256*DDIM, 0, 3);
    __builtin_prefetch(cl0 + 256*DDIM, 0, 3);
    __builtin_prefetch(ch1 + 256*DDIM, 0, 3);
    __builtin_prefetch(cl1 + 256*DDIM, 0, 3);

    v8f acc0 = {0.f,0.f,0.f,0.f,0.f,0.f,0.f,0.f};
    v8f acc1 = {0.f,0.f,0.f,0.f,0.f,0.f,0.f,0.f};
#pragma unroll
    for(int c = 0; c < 4; ++c){
      // 16-bit B layout: lane(0-15)=N holds K 0-7,16-23; lane(16-31)=N holds K 8-15,24-31.
      const int k0 = 32*c + 8*half;
      Frag bh0, bl0, bh1, bl1;
      bh0.q[0] = *(const uint4*)(ch0 + k0);
      bh0.q[1] = *(const uint4*)(ch0 + k0 + 16);
      bl0.q[0] = *(const uint4*)(cl0 + k0);
      bl0.q[1] = *(const uint4*)(cl0 + k0 + 16);
      bh1.q[0] = *(const uint4*)(ch1 + k0);
      bh1.q[1] = *(const uint4*)(ch1 + k0 + 16);
      bl1.q[0] = *(const uint4*)(cl1 + k0);
      bl1.q[1] = *(const uint4*)(cl1 + k0 + 16);
      acc0 = __builtin_amdgcn_wmma_f32_16x16x32_bf16(false, ahi[c], false, bh0.v, (short)0, acc0, false, false);
      acc0 = __builtin_amdgcn_wmma_f32_16x16x32_bf16(false, ahi[c], false, bl0.v, (short)0, acc0, false, false);
      acc0 = __builtin_amdgcn_wmma_f32_16x16x32_bf16(false, alo[c], false, bh0.v, (short)0, acc0, false, false);
      acc1 = __builtin_amdgcn_wmma_f32_16x16x32_bf16(false, ahi[c], false, bh1.v, (short)0, acc1, false, false);
      acc1 = __builtin_amdgcn_wmma_f32_16x16x32_bf16(false, ahi[c], false, bl1.v, (short)0, acc1, false, false);
      acc1 = __builtin_amdgcn_wmma_f32_16x16x32_bf16(false, alo[c], false, bh1.v, (short)0, acc1, false, false);
    }
    // C layout: lane n VGPR i = (M=i, N=n); lane n+16 VGPR i = (M=8+i, N=n).
    // Two logits per row per iteration -> amortized online update.
#pragma unroll
    for(int i = 0; i < 8; ++i){
      float v1 = acc0[i], v2 = acc1[i];
      float tm = fmaxf(v1, v2);
      float nm = fmaxf(m[i], tm);
      float e12 = ex2(v1 - nm) + ex2(v2 - nm);
      s[i] = fmaf(s[i], ex2(m[i] - nm), e12);
      m[i] = nm;
    }
  }

  // reduce (m,s) across the 16 lanes of each half (rows 0-7 / 8-15)
#pragma unroll
  for(int i = 0; i < 8; ++i){
#pragma unroll
    for(int off = 1; off < 16; off <<= 1){
      float om = __shfl_xor(m[i], off, 32);
      float os = __shfl_xor(s[i], off, 32);
      float nm = fmaxf(m[i], om);
      s[i] = s[i]*ex2(m[i]-nm) + os*ex2(om-nm);
      m[i] = nm;
    }
  }

  __shared__ float sm[8][16];
  __shared__ float ss[8][16];
  if(l16 == 0){
#pragma unroll
    for(int i = 0; i < 8; ++i){
      sm[wave][half*8 + i] = m[i];
      ss[wave][half*8 + i] = s[i];
    }
  }
  __syncthreads();

  if(threadIdx.x < 16){
    const int t = threadIdx.x;
    float M = -3.0e38f, S = 0.f;
#pragma unroll
    for(int w = 0; w < 8; ++w){
      float om = sm[w][t], os = ss[w][t];
      float nm = fmaxf(M, om);
      S = S*ex2(M-nm) + os*ex2(om-nm);
      M = nm;
    }
    const float lse2 = M + lg2(S);           // log2(sum_j 2^(log2e*logit))
    const int e = r0 + t;
    const float4* pe = (const float4*)(parts   + (size_t)e * DDIM);
    const float4* ce = (const float4*)(centers + (size_t)e * DDIM);
    float diag = 0.f;                        // exact fp32 diagonal logit
#pragma unroll
    for(int j = 0; j < DDIM/4; ++j){
      float4 a = pe[j], b = ce[j];
      diag = fmaf(a.x, b.x, diag); diag = fmaf(a.y, b.y, diag);
      diag = fmaf(a.z, b.z, diag); diag = fmaf(a.w, b.w, diag);
    }
    row_loss[e] = 0.69314718055994531f * lse2 - diag;   // lse - logits[e,e]
  }
}

// deterministic tree reduction -> scalar mean
__global__ void sce_reduce(const float* __restrict__ row_loss, float* __restrict__ out){
  __shared__ float buf[256];
  float s = 0.f;
  for(int i = threadIdx.x; i < KROWS; i += 256) s += row_loss[i];
  buf[threadIdx.x] = s;
  __syncthreads();
  for(int o = 128; o > 0; o >>= 1){
    if(threadIdx.x < o) buf[threadIdx.x] += buf[threadIdx.x + o];
    __syncthreads();
  }
  if(threadIdx.x == 0) out[0] = buf[0] * (1.0f / KROWS);
}

extern "C" void kernel_launch(void* const* d_in, const int* in_sizes, int n_in,
                              void* d_out, int out_size, void* d_ws, size_t ws_size,
                              hipStream_t stream){
  const float* parts   = (const float*)d_in[0];
  const float* centers = (const float*)d_in[1];
  char* ws = (char*)d_ws;
  const size_t KD = (size_t)KROWS * DDIM;     // 2,097,152 elements

  bf16*  Phi      = (bf16*)(ws);
  bf16*  Plo      = (bf16*)(ws + 2*KD);
  bf16*  Chi      = (bf16*)(ws + 4*KD);
  bf16*  Clo      = (bf16*)(ws + 6*KD);
  float* row_loss = (float*)(ws + 8*KD);      // total ws use ~16.8 MB

  const int n = (int)KD;
  const float LOG2E = 1.4426950408889634f;    // fold base-2 scaling into parts
  sce_convert<<<(n + 255)/256, 256, 0, stream>>>(parts,   Phi, Plo, LOG2E, n);
  sce_convert<<<(n + 255)/256, 256, 0, stream>>>(centers, Chi, Clo, 1.0f,  n);
  sce_main<<<KROWS/16, 256, 0, stream>>>(Phi, Plo, Chi, Clo, parts, centers, row_loss);
  sce_reduce<<<1, 256, 0, stream>>>(row_loss, (float*)d_out);
}
